// HungarianMatcher_32830730011570
// MI455X (gfx1250) — matrix operand order repaired and checked by reading, not compiled, not run on hardware
//
#include <hip/hip_runtime.h>

typedef __attribute__((ext_vector_type(16))) _Float16 v16h;
typedef __attribute__((ext_vector_type(8)))  _Float16 v8h;
typedef __attribute__((ext_vector_type(4)))  _Float16 v4h;
typedef __attribute__((ext_vector_type(8)))  float    v8f;

#define B_      4
#define Q_      256
#define E_      128
#define HW_     36864
#define SPLITS  64
#define KCHUNK  (HW_ / SPLITS)    // 576
#define KSTEP   32
#define KITERS  (KCHUNK / KSTEP)  // 18
#define MT      128               // q-rows per block (8 waves x 16)
#define ROWH    40                // halves per LDS row: 32 data + 8 pad (80B, 16B aligned stride)

__device__ __forceinline__ float softplus_f(float x) {
    // stable: max(x,0) + log(1 + exp(-|x|))
    return fmaxf(x, 0.0f) + __logf(1.0f + __expf(-fabsf(x)));
}

__device__ __forceinline__ v16h cat16(v8h lo, v8h hi) {
    return __builtin_shufflevector(lo, hi, 0,1,2,3,4,5,6,7,8,9,10,11,12,13,14,15);
}

// ---------------------------------------------------------------------------
// Split-K fused GEMM:
//   G1[b,q,e] += sum_k (-mask*pred)[b,q,k] * tgt[b,e,k]
//   G2[b,q,e] += sum_k (sigmoid(portion)*pred)[b,q,k] * segmap[b,e,k]
// plus row sums negsum[b,q], psum[b,q], segsum[b,e], nnz[b].
// grid = (SPLITS, Q_/MT, B_), block = 256 threads (8 wave32).
// ---------------------------------------------------------------------------
__global__ __launch_bounds__(256, 1)
void cost_gemm_kernel(const float* __restrict__ mask_logits,
                      const float* __restrict__ pred_mask,
                      const float* __restrict__ portion_logits,
                      const float* __restrict__ segmap,
                      float* __restrict__ G1, float* __restrict__ G2,
                      float* __restrict__ negsum, float* __restrict__ psum,
                      float* __restrict__ segsum, float* __restrict__ nnz)
{
    __shared__ __align__(16) _Float16 sA1[MT * ROWH];  // (-x*pred) f16, rows=q
    __shared__ __align__(16) _Float16 sA2[MT * ROWH];  // sigmoid(portion)*pred
    __shared__ __align__(16) _Float16 sBt[E_ * ROWH];  // tgt binary, rows=e
    __shared__ __align__(16) _Float16 sBs[E_ * ROWH];  // segmap
    __shared__ float sNeg[MT];
    __shared__ float sP[MT];
    __shared__ float sSeg[E_];
    __shared__ float sNnz;

    const int t     = threadIdx.x;
    const int b     = blockIdx.z;
    const int qt    = blockIdx.y;
    const int q0    = qt * MT;
    const int kbase = blockIdx.x * KCHUNK;

    if (t < MT) { sNeg[t] = 0.0f; sP[t] = 0.0f; sSeg[t] = 0.0f; }
    if (t == 0) sNnz = 0.0f;

    // load mapping: thread t handles float4 column c4 = t&7 of rows r0+32j
    const int c4 = t & 7;
    const int r0 = t >> 3;

    const int lane = t & 31;
    const int wave = t >> 5;
    const int n    = lane & 15;   // N column (B/C/D) == M row (A) for this lane
    const int half = lane >> 4;

    const float* mb = mask_logits    + (size_t)(b * Q_ + q0) * HW_;
    const float* pb = pred_mask      + (size_t)(b * Q_ + q0) * HW_;
    const float* ob = portion_logits + (size_t)(b * Q_ + q0) * HW_;
    const float* sb = segmap         + (size_t)b * E_ * HW_;

    v8f acc1[8] = {};
    v8f acc2[8] = {};

    float negL[4]  = {0.f, 0.f, 0.f, 0.f};
    float psumL[4] = {0.f, 0.f, 0.f, 0.f};
    float segL[4]  = {0.f, 0.f, 0.f, 0.f};
    float nnzL = 0.0f;

    for (int s = 0; s < KITERS; ++s) {
        const int k0 = kbase + s * KSTEP;
        __syncthreads();  // LDS reads of previous tile complete

        // ---- load + fused transform into LDS (f32 -> f16) ----
#pragma unroll
        for (int j = 0; j < 4; ++j) {
            const int row  = r0 + 32 * j;                       // q-row / e-row in tile
            const size_t g = (size_t)row * HW_ + k0 + c4 * 4;
            const float4 x4 = *(const float4*)(mb + g);
            const float4 p4 = *(const float4*)(pb + g);
            const float4 o4 = *(const float4*)(ob + g);
            const float4 s4 = *(const float4*)(sb + g);         // row < E_ == MT

            const float xs[4] = {x4.x, x4.y, x4.z, x4.w};
            const float ps[4] = {p4.x, p4.y, p4.z, p4.w};
            const float os[4] = {o4.x, o4.y, o4.z, o4.w};
            const float ss[4] = {s4.x, s4.y, s4.z, s4.w};

            v4h a1, a2, bt, bs;
            float negA = 0.f, pA = 0.f, segA = 0.f, nnzA = 0.f;
#pragma unroll
            for (int c = 0; c < 4; ++c) {
                const float x  = xs[c];
                const float pm = ps[c];
                const float sg = ss[c];
                const float sig = 1.0f / (1.0f + __expf(-os[c]));
                const float pv  = sig * pm;
                const float tg  = (sg > 0.0f) ? 1.0f : 0.0f;
                a1[c] = (_Float16)(-x * pm);
                a2[c] = (_Float16)pv;
                bt[c] = (_Float16)tg;
                bs[c] = (_Float16)sg;
                negA += softplus_f(x) * pm;
                pA   += pv;
                segA += sg;
                nnzA += tg;
            }
            *(v4h*)(sA1 + row * ROWH + c4 * 4) = a1;
            *(v4h*)(sA2 + row * ROWH + c4 * 4) = a2;
            *(v4h*)(sBt + row * ROWH + c4 * 4) = bt;
            *(v4h*)(sBs + row * ROWH + c4 * 4) = bs;
            negL[j]  += negA;
            psumL[j] += pA;
            segL[j]  += segA;
            nnzL     += nnzA;
        }
        __syncthreads();

        // ---- WMMA phase: wave owns 16 q-rows x full 128 e-cols ----
        // A fragment (16x32 f16): lane(m=n, half): V0..7 -> K=half*16+0..15
        const _Float16* aRow1 = sA1 + (wave * 16 + n) * ROWH;
        const _Float16* aRow2 = sA2 + (wave * 16 + n) * ROWH;
        const v16h A1 = cat16(*(const v8h*)(aRow1 + half * 16),
                              *(const v8h*)(aRow1 + half * 16 + 8));
        const v16h A2 = cat16(*(const v8h*)(aRow2 + half * 16),
                              *(const v8h*)(aRow2 + half * 16 + 8));
#pragma unroll
        for (int j = 0; j < 8; ++j) {
            // B fragment (32x16 f16): lane(col=n, half): V0..7 -> K=half*16+0..15
            const _Float16* btp = sBt + (j * 16 + n) * ROWH + half * 16;
            const _Float16* bsp = sBs + (j * 16 + n) * ROWH + half * 16;
            const v16h Bt = cat16(*(const v8h*)btp, *(const v8h*)(btp + 8));
            const v16h Bs = cat16(*(const v8h*)bsp, *(const v8h*)(bsp + 8));
            acc1[j] = __builtin_amdgcn_wmma_f32_16x16x32_f16(
                false, A1, false, Bt, (short)0, acc1[j], false, false);
            acc2[j] = __builtin_amdgcn_wmma_f32_16x16x32_f16(
                false, A2, false, Bs, (short)0, acc2[j], false, false);
        }
    }

    // ---- row-sum reduction ----
    __syncthreads();
#pragma unroll
    for (int j = 0; j < 4; ++j) {
        const int row = r0 + 32 * j;
        atomicAdd(&sNeg[row], negL[j]);
        atomicAdd(&sP[row],   psumL[j]);
        atomicAdd(&sSeg[row], segL[j]);
    }
    atomicAdd(&sNnz, nnzL);
    __syncthreads();
    if (t < MT) {
        atomicAdd(&negsum[b * Q_ + q0 + t], sNeg[t]);
        atomicAdd(&psum[b * Q_ + q0 + t],   sP[t]);
        if (qt == 0) atomicAdd(&segsum[b * E_ + t], sSeg[t]);  // B tile shared across q-tiles
    }
    if (t == 0 && qt == 0) atomicAdd(&nnz[b], sNnz);

    // ---- accumulate partial products ----
    // C/D layout: VGPR v, lane(n,half) -> (M = half*8 + v, N = n)
#pragma unroll
    for (int j = 0; j < 8; ++j) {
#pragma unroll
        for (int v = 0; v < 8; ++v) {
            const int q = q0 + wave * 16 + half * 8 + v;
            const int e = j * 16 + n;
            const size_t o = ((size_t)(b * Q_ + q)) * E_ + e;
            atomicAdd(&G1[o], acc1[j][v]);
            atomicAdd(&G2[o], acc2[j][v]);
        }
    }
}

// ---------------------------------------------------------------------------
// Final combine: class softplus + BCE + dice + huber  ->  out[B,Q,E]
// ---------------------------------------------------------------------------
__global__ __launch_bounds__(256)
void cost_combine_kernel(const float* __restrict__ G1, const float* __restrict__ G2,
                         const float* __restrict__ negsum, const float* __restrict__ psum,
                         const float* __restrict__ segsum, const float* __restrict__ nnz,
                         const float* __restrict__ positions,
                         const float* __restrict__ true_positions,
                         const float* __restrict__ is_e,
                         float* __restrict__ out)
{
    const int gid = blockIdx.x * 256 + threadIdx.x;
    if (gid >= B_ * Q_ * E_) return;
    const int e = gid % E_;
    const int q = (gid / E_) % Q_;
    const int b = gid / (E_ * Q_);

    const float cls  = softplus_f(-is_e[b * Q_ + q]);
    const float bce  = (negsum[b * Q_ + q] + G1[gid]) / nnz[b];
    const float dice = 1.0f - (2.0f * G2[gid] + 1.0f)
                             / (psum[b * Q_ + q] + segsum[b * E_ + e] + 1.0f);
    float dist = 0.0f;
#pragma unroll
    for (int d = 0; d < 2; ++d) {
        const float df = positions[(b * Q_ + q) * 2 + d]
                       - true_positions[(b * E_ + e) * 2 + d];
        const float a = fabsf(df);
        dist += (a < 1.0f) ? (0.5f * a * a) : (a - 0.5f);
    }
    dist *= 0.5f;  // mean over last dim (size 2)

    out[gid] = cls + bce + dice + dist;
}

extern "C" void kernel_launch(void* const* d_in, const int* in_sizes, int n_in,
                              void* d_out, int out_size, void* d_ws, size_t ws_size,
                              hipStream_t stream) {
    (void)in_sizes; (void)n_in; (void)out_size; (void)ws_size;

    const float* mask_logits    = (const float*)d_in[0];
    const float* pred_mask      = (const float*)d_in[1];
    const float* portion_logits = (const float*)d_in[2];
    const float* segmap         = (const float*)d_in[3];
    const float* positions      = (const float*)d_in[4];
    const float* true_positions = (const float*)d_in[5];
    const float* is_e           = (const float*)d_in[6];

    float* ws = (float*)d_ws;
    float* G1     = ws;                                 // B*Q*E
    float* G2     = G1 + (size_t)B_ * Q_ * E_;          // B*Q*E
    float* negsum = G2 + (size_t)B_ * Q_ * E_;          // B*Q
    float* psum   = negsum + B_ * Q_;                   // B*Q
    float* segsum = psum + B_ * Q_;                     // B*E
    float* nnzp   = segsum + B_ * E_;                   // B
    const size_t ws_floats = (size_t)2 * B_ * Q_ * E_ + 2 * B_ * Q_ + B_ * E_ + B_;

    hipMemsetAsync(d_ws, 0, ws_floats * sizeof(float), stream);

    dim3 grid(SPLITS, Q_ / MT, B_);
    cost_gemm_kernel<<<grid, 256, 0, stream>>>(
        mask_logits, pred_mask, portion_logits, segmap,
        G1, G2, negsum, psum, segsum, nnzp);

    const int total = B_ * Q_ * E_;
    cost_combine_kernel<<<(total + 255) / 256, 256, 0, stream>>>(
        G1, G2, negsum, psum, segsum, nnzp,
        positions, true_positions, is_e, (float*)d_out);
}
